// BSDENet_11519102288370
// MI455X (gfx1250) — matrix operand order repaired and checked by reading, not compiled, not run on hardware
//
#include <hip/hip_runtime.h>

#define NUM_WG   16
#define BLOCK    256
#define BATCH    2048
#define DIMS     64
#define TSTEPS   100
#define HSTEP    0.01f

typedef __attribute__((ext_vector_type(16))) _Float16 v16h;
typedef __attribute__((ext_vector_type(8)))  _Float16 v8h;
typedef __attribute__((ext_vector_type(8)))  float    v8f;

// ---------------- LDS layout (bytes) ----------------
#define SM_ACT    0u        // 128x128 f16 activations      32768
#define SM_W1T    32768u    // layer1 weights^T f16         24576 (z: 128x96, y: 128x64)
#define SM_W2T    57344u    // layer2 weights^T f16 128x128 32768
#define SM_W3T    90112u    // z layer3 weights^T f16 64x128 16384
#define SM_XS     106496u   // x state f32 128x64           32768
#define SM_YS     139264u   // y state f32 128               512
#define SM_SUMSP  139776u   // per-wave col-sum partials 8x128 f32 4096
#define SM_SUMQP  143872u   // per-wave col-sumsq partials         4096
#define SM_MEAN   147968u   // 512
#define SM_ISTD   148480u   // 512
#define SM_G1     149504u
#define SM_BE1    150016u
#define SM_G2     151040u
#define SM_BE2    151552u
#define SM_B3     152064u
#define SM_ROWA   152576u   // per-row zz / loss partials
#define SM_ROWB   153088u   // per-row z.dW partials
#define SM_YW3    153600u   // y0 final weights f32 128
#define SM_YB3    154112u
#define LDS_BYTES 154624u

struct KernelArgs {
  const float* x; const float* bm;
  const float *zg1,*zbe1,*zg2,*zbe2,*zb3;
  const float *yg1,*ybe1,*yg2,*ybe2,*yW3v,*yb3;
  const unsigned short *zW1t,*zW2t,*zW3t,*yW1t,*yW2t;
  float* wsBN; float* wsLoss; unsigned* bar; float* out;
};

struct PrepArgs {
  const float *zW1,*zW2,*zW3,*yW1,*yW2;
  unsigned short *zW1t,*zW2t,*zW3t,*yW1t,*yW2t;
  unsigned* bar;
};

// ---------------- device helpers ----------------

__device__ __forceinline__ void grid_barrier(unsigned* bar) {
  __syncthreads();
  if (threadIdx.x == 0) {
    __threadfence();
    volatile unsigned* gen = bar + 1;
    unsigned g = *gen;
    unsigned arrived = atomicAdd(bar, 1u);
    if (arrived == NUM_WG - 1) {
      bar[0] = 0u;
      __threadfence();
      *gen = g + 1u;
    } else {
      while (*gen == g) { __builtin_amdgcn_s_sleep(2); }
    }
    __threadfence();
  }
  __syncthreads();
}

// Async global->LDS copy (GVS mode: per-lane LDS offset VGPR, 32-bit offset VGPR,
// SGPR base pair). All 256 threads participate; nbytes must be a multiple of 4096.
__device__ __forceinline__ void async_copy_g2l(unsigned lds_byte_off, const void* gbase, int nbytes) {
  const int tid = threadIdx.x;
  for (int o = tid * 16; o < nbytes; o += BLOCK * 16) {
    unsigned ldsa = lds_byte_off + (unsigned)o;
    unsigned goff = (unsigned)o;
    asm volatile("global_load_async_to_lds_b128 %0, %1, %2 offset:0"
                 :: "v"(ldsa), "v"(goff), "s"(gbase)
                 : "memory");
  }
  asm volatile("s_wait_asynccnt 0x0" ::: "memory");
}

// A fragment (16x32 f16): lanes 0-15 hold M=lane, K=kb..kb+15;
// lanes 16-31 hold M=lane-16, K=kb+16..kb+31 (contiguous).
__device__ __forceinline__ v16h load_afrag(const _Float16* act, int mr, int kb) {
  const int lane = threadIdx.x & 31;
  const int hi = lane >> 4, r = lane & 15;
  const _Float16* base = act + (mr + r) * 128 + kb + 16 * hi;
  v8h lo = *(const v8h*)base;
  v8h hh = *(const v8h*)(base + 8);
  v16h a;
#pragma unroll
  for (int i = 0; i < 8; ++i) { a[i] = lo[i]; a[i + 8] = hh[i]; }
  return a;
}

// B fragment (32x16 f16) from transposed weights Wt[N][Kpad]:
// lane holds column N = nb + (lane&15), K = kb + 16*(lane>>4) .. +15 (contiguous).
__device__ __forceinline__ v16h load_bfrag(const _Float16* wT, int wStride, int nb, int kb) {
  const int lane = threadIdx.x & 31;
  const int hi = lane >> 4, r = lane & 15;
  return *(const v16h*)(wT + (nb + r) * wStride + kb + 16 * hi);
}

// Linear (no bias: absorbed by BN) + BatchNorm(batch stats over 2048) + ReLU.
// in/out = act (f16, stride 128). Templated so K-loop fully unrolls and the
// first WMMA folds its C operand to inline 0.
template <int KC, int WS>
__device__ __forceinline__ void dense_bn_relu(
    _Float16* act, const _Float16* wT,
    const float* gam, const float* bet,
    float* sumsP, float* sumqP, float* meanL, float* istdL,
    float* wsBN, unsigned* bar, int wg)
{
  const int tid  = threadIdx.x;
  const int lane = tid & 31, wave = tid >> 5;
  const int mr = wave * 16, cl = lane & 15, hi = lane >> 4;

  __syncthreads();                       // act writes from caller visible

  v8f acc[8];
#pragma unroll
  for (int n = 0; n < 8; ++n)
#pragma unroll
    for (int v = 0; v < 8; ++v) acc[n][v] = 0.0f;

#pragma unroll
  for (int kc = 0; kc < KC; ++kc) {
    v16h a = load_afrag(act, mr, kc * 32);
#pragma unroll
    for (int n = 0; n < 8; ++n) {
      v16h b = load_bfrag(wT, WS, n * 16, kc * 32);
      acc[n] = __builtin_amdgcn_wmma_f32_16x16x32_f16(
          false, a, false, b, (short)0, acc[n], false, false);
    }
  }
  // deterministic column stats: register -> cross-half shuffle -> per-wave LDS partials
#pragma unroll
  for (int n = 0; n < 8; ++n) {
    float s = 0.f, q = 0.f;
#pragma unroll
    for (int v = 0; v < 8; ++v) { float xv = acc[n][v]; s += xv; q += xv * xv; }
    s += __shfl_xor(s, 16, 32);
    q += __shfl_xor(q, 16, 32);
    if (hi == 0) { sumsP[wave * 128 + n * 16 + cl] = s; sumqP[wave * 128 + n * 16 + cl] = q; }
  }
  __syncthreads();
  if (tid < 128) {
    float S = 0.f, Q = 0.f;
#pragma unroll
    for (int w = 0; w < 8; ++w) { S += sumsP[w * 128 + tid]; Q += sumqP[w * 128 + tid]; }
    wsBN[wg * 256 + tid] = S;
    wsBN[wg * 256 + 128 + tid] = Q;
  }
  grid_barrier(bar);
  if (tid < 128) {
    float S = 0.f, Q = 0.f;
    for (int s = 0; s < NUM_WG; ++s) { S += wsBN[s * 256 + tid]; Q += wsBN[s * 256 + 128 + tid]; }
    float m = S * (1.f / (float)BATCH);
    float var = Q * (1.f / (float)BATCH) - m * m;
    meanL[tid] = m;
    istdL[tid] = rsqrtf(var + 1e-5f);
  }
  __syncthreads();
#pragma unroll
  for (int n = 0; n < 8; ++n) {
    int c = n * 16 + cl;
    float m = meanL[c], is = istdL[c], g = gam[c], be = bet[c];
#pragma unroll
    for (int v = 0; v < 8; ++v) {
      float xv = (acc[n][v] - m) * is * g + be;
      xv = fmaxf(xv, 0.f);
      act[(mr + v + 8 * hi) * 128 + c] = (_Float16)xv;
    }
  }
  __syncthreads();
}

// ---------------- kernels ----------------

// Convert f32 weights to f16, transposed to [N][Kpad]; also reset the grid barrier.
__global__ __launch_bounds__(BLOCK) void bsde_prep(PrepArgs pp) {
  int i = blockIdx.x * BLOCK + threadIdx.x;
  if (i < 2) pp.bar[i] = 0u;
  _Float16* zW1t = (_Float16*)pp.zW1t;
  _Float16* zW2t = (_Float16*)pp.zW2t;
  _Float16* zW3t = (_Float16*)pp.zW3t;
  _Float16* yW1t = (_Float16*)pp.yW1t;
  _Float16* yW2t = (_Float16*)pp.yW2t;
  if (i < 12288) {                       // zW1t [128][96], K padded 66->96
    int n = i / 96, k = i - n * 96;
    zW1t[i] = (_Float16)((k < 66) ? pp.zW1[k * 128 + n] : 0.0f);
  } else if (i < 28672) {                // zW2t [128][128]
    int j = i - 12288; int n = j >> 7, k = j & 127;
    zW2t[j] = (_Float16)pp.zW2[k * 128 + n];
  } else if (i < 36864) {                // zW3t [64][128]
    int j = i - 28672; int n = j >> 7, k = j & 127;
    zW3t[j] = (_Float16)pp.zW3[k * 64 + n];
  } else if (i < 45056) {                // yW1t [128][64]
    int j = i - 36864; int n = j >> 6, k = j & 63;
    yW1t[j] = (_Float16)pp.yW1[k * 128 + n];
  } else if (i < 61440) {                // yW2t [128][128]
    int j = i - 45056; int n = j >> 7, k = j & 127;
    yW2t[j] = (_Float16)pp.yW2[k * 128 + n];
  }
}

__global__ __launch_bounds__(BLOCK) void bsde_main(KernelArgs p) {
  extern __shared__ char sm[];
  _Float16* act  = (_Float16*)(sm + SM_ACT);
  _Float16* w1t  = (_Float16*)(sm + SM_W1T);
  _Float16* w2t  = (_Float16*)(sm + SM_W2T);
  _Float16* w3t  = (_Float16*)(sm + SM_W3T);
  float* xs    = (float*)(sm + SM_XS);
  float* ys    = (float*)(sm + SM_YS);
  float* sumsP = (float*)(sm + SM_SUMSP);
  float* sumqP = (float*)(sm + SM_SUMQP);
  float* meanL = (float*)(sm + SM_MEAN);
  float* istdL = (float*)(sm + SM_ISTD);
  float* g1  = (float*)(sm + SM_G1);
  float* be1 = (float*)(sm + SM_BE1);
  float* g2  = (float*)(sm + SM_G2);
  float* be2 = (float*)(sm + SM_BE2);
  float* b3L = (float*)(sm + SM_B3);
  float* rowA = (float*)(sm + SM_ROWA);
  float* rowB = (float*)(sm + SM_ROWB);
  float* yw3 = (float*)(sm + SM_YW3);
  float* yb3 = (float*)(sm + SM_YB3);

  const int tid = threadIdx.x;
  const int wg  = blockIdx.x;
  const int lane = tid & 31, wave = tid >> 5;
  const int mr = wave * 16, cl = lane & 15, hi = lane >> 4;

  // -------- load x tile + y0-net parameters --------
  for (int i = tid; i < 128 * 64; i += BLOCK) xs[i] = p.x[wg * 128 * 64 + i];
  if (tid < 128) {
    g1[tid] = p.yg1[tid]; be1[tid] = p.ybe1[tid];
    g2[tid] = p.yg2[tid]; be2[tid] = p.ybe2[tid];
    yw3[tid] = p.yW3v[tid];
  }
  if (tid == 0) yb3[0] = p.yb3[0];
  async_copy_g2l(SM_W1T, p.yW1t, 128 * 64 * 2);
  async_copy_g2l(SM_W2T, p.yW2t, 128 * 128 * 2);
  __syncthreads();

  // -------- y = y0_mlp(x) --------
  for (int i = tid; i < 128 * 64; i += BLOCK) {
    int r = i >> 6, c = i & 63;
    act[r * 128 + c] = (_Float16)xs[i];
  }
  dense_bn_relu<2, 64>(act, w1t, g1, be1, sumsP, sumqP, meanL, istdL, p.wsBN, p.bar, wg);
  dense_bn_relu<4, 128>(act, w2t, g2, be2, sumsP, sumqP, meanL, istdL, p.wsBN, p.bar, wg);
  if (tid < 128) {                       // final Linear (128 -> 1) via VALU dot
    float s = yb3[0];
    for (int k = 0; k < 128; ++k) s += (float)act[tid * 128 + k] * yw3[k];
    ys[tid] = s;
  }
  __syncthreads();

  // -------- swap in z-net parameters --------
  if (tid < 128) {
    g1[tid] = p.zg1[tid]; be1[tid] = p.zbe1[tid];
    g2[tid] = p.zg2[tid]; be2[tid] = p.zbe2[tid];
  }
  if (tid < 64) b3L[tid] = p.zb3[tid];
  async_copy_g2l(SM_W1T, p.zW1t, 128 * 96 * 2);
  async_copy_g2l(SM_W2T, p.zW2t, 128 * 128 * 2);
  async_copy_g2l(SM_W3T, p.zW3t, 64 * 128 * 2);
  __syncthreads();

  float b3v[4];
#pragma unroll
  for (int n = 0; n < 4; ++n) b3v[n] = b3L[n * 16 + cl];

  // -------- Euler-Maruyama scan (steps 0..99) + final eval (step 100, t=1) --------
#pragma unroll 1
  for (int step = 0; step <= TSTEPS; ++step) {
    const bool fin = (step == TSTEPS);
    const float tval = fin ? 1.0f : (float)step * HSTEP;

    // prefetch this step's Brownian increments (only HBM-latency reads in the loop)
    if (!fin) {
      const char* pb = (const char*)(p.bm + ((size_t)(wg * 128) * TSTEPS + step) * 64);
      if (tid < 128) {
        const char* q = pb + (size_t)tid * TSTEPS * 64 * 4;
        __builtin_prefetch(q, 0, 1);
        __builtin_prefetch(q + 128, 0, 1);
      }
    }

    // build A = [x, t, y, 0-pad] as f16, K padded 66 -> 96
    for (int i = tid; i < 128 * 96; i += BLOCK) {
      int r = i / 96, c = i - r * 96;
      float v;
      if (c < 64)      v = xs[r * 64 + c];
      else if (c == 64) v = tval;
      else if (c == 65) v = ys[r];
      else              v = 0.f;
      act[r * 128 + c] = (_Float16)v;
    }
    dense_bn_relu<3, 96>(act, w1t, g1, be1, sumsP, sumqP, meanL, istdL, p.wsBN, p.bar, wg);
    dense_bn_relu<4, 128>(act, w2t, g2, be2, sumsP, sumqP, meanL, istdL, p.wsBN, p.bar, wg);

    // layer 3: z[128x64] = act @ W3 + b3 (no BN); bias added post-GEMM
    v8f acc[4];
#pragma unroll
    for (int n = 0; n < 4; ++n)
#pragma unroll
      for (int v = 0; v < 8; ++v) acc[n][v] = 0.0f;
#pragma unroll
    for (int kc = 0; kc < 4; ++kc) {
      v16h a = load_afrag(act, mr, kc * 32);
#pragma unroll
      for (int n = 0; n < 4; ++n) {
        v16h b = load_bfrag(w3t, 128, n * 16, kc * 32);
        acc[n] = __builtin_amdgcn_wmma_f32_16x16x32_f16(
            false, a, false, b, (short)0, acc[n], false, false);
      }
    }

    if (!fin) {
      // x/y update: each lane owns 8 rows x 4 cols of z (C-layout row = mr+v+8*hi, col = n*16+cl)
#pragma unroll
      for (int v = 0; v < 8; ++v) {
        int r = mr + v + 8 * hi;
        int gb = wg * 128 + r;
        float szz = 0.f, szdw = 0.f;
#pragma unroll
        for (int n = 0; n < 4; ++n) {
          int c = n * 16 + cl;
          float z  = acc[n][v] + b3v[n];
          float dw = p.bm[(gb * TSTEPS + step) * 64 + c];
          float xc = xs[r * 64 + c];
          float xn = xc + (0.5f * xc + z) * HSTEP + dw;
          xs[r * 64 + c] = (fabsf(xn) > 20.f) ? 0.f : xn;
          szz  += z * z;
          szdw += z * dw;
        }
#pragma unroll
        for (int m = 1; m < 16; m <<= 1) {
          szz  += __shfl_xor(szz,  m, 32);
          szdw += __shfl_xor(szdw, m, 32);
        }
        if (cl == 0) { rowA[r] = szz; rowB[r] = szdw; }
      }
      __syncthreads();
      if (tid < 128) ys[tid] += (-0.5f + 0.5f * rowA[tid]) * HSTEP + rowB[tid];
      __syncthreads();
    } else {
      // loss: mean_B( sum_D (-x - z)^2 )
#pragma unroll
      for (int v = 0; v < 8; ++v) {
        int r = mr + v + 8 * hi;
        float sd = 0.f;
#pragma unroll
        for (int n = 0; n < 4; ++n) {
          int c = n * 16 + cl;
          float d = -xs[r * 64 + c] - (acc[n][v] + b3v[n]);
          sd += d * d;
        }
#pragma unroll
        for (int m = 1; m < 16; m <<= 1) sd += __shfl_xor(sd, m, 32);
        if (cl == 0) rowA[r] = sd;
      }
      __syncthreads();
      if (tid == 0) {
        float s = 0.f;
        for (int r = 0; r < 128; ++r) s += rowA[r];
        p.wsLoss[wg] = s;
      }
      grid_barrier(p.bar);
      if (wg == 0 && tid == 0) {
        float s = 0.f;
        for (int i = 0; i < NUM_WG; ++i) s += p.wsLoss[i];
        p.out[0] = s * (1.f / (float)BATCH);
      }
    }
  }
}

// ---------------- host ----------------

extern "C" void kernel_launch(void* const* d_in, const int* in_sizes, int n_in,
                              void* d_out, int out_size, void* d_ws, size_t ws_size,
                              hipStream_t stream) {
  (void)out_size; (void)ws_size;
  // Locate inputs robustly against pytree flattening order (insertion vs sorted keys).
  int ix = -1, ibm = -1, rem[32], nr = 0;
  for (int i = 0; i < n_in; ++i) {
    if (in_sizes[i] == BATCH * DIMS && ix < 0) ix = i;
    else if (in_sizes[i] == BATCH * TSTEPS * DIMS && ibm < 0) ibm = i;
    else if (nr < 32) rem[nr++] = i;
  }
  if (ix < 0 || ibm < 0 || nr < 20) return;
  int p8448 = -1;
  for (int j = 0; j < 20; ++j) if (in_sizes[rem[j]] == 8448) { p8448 = j; break; }
  const bool zfirst = (p8448 >= 0 && p8448 < 10);
  int zb[10], yb[10];
  for (int j = 0; j < 10; ++j) { zb[j] = rem[zfirst ? j : 10 + j]; yb[j] = rem[zfirst ? 10 + j : j]; }
  // canonical order: W1,b1,W2,b2,W3,b3,g1,be1,g2,be2
  auto canon = [&](const int* b, int w1sz, int* c) {
    if (in_sizes[b[0]] == w1sz) { for (int j = 0; j < 10; ++j) c[j] = b[j]; }
    else { const int map[10] = {4,5,6,7,8,9,0,1,2,3}; for (int j = 0; j < 10; ++j) c[j] = b[map[j]]; }
  };
  int zc[10], yc[10];
  canon(zb, 8448, zc);
  canon(yb, 8192, yc);

  // workspace carve
  char* w = (char*)d_ws;
  unsigned* bar   = (unsigned*)w;                 // 2 words (reset by prep kernel)
  float* wsLoss   = (float*)(w + 64);             // 16 floats
  float* wsBN     = (float*)(w + 128);            // 16 WGs x 256 floats = 16 KB
  unsigned short* zW1t = (unsigned short*)(w + 16640);   // 128*96
  unsigned short* zW2t = (unsigned short*)(w + 41216);   // 128*128
  unsigned short* zW3t = (unsigned short*)(w + 73984);   // 64*128
  unsigned short* yW1t = (unsigned short*)(w + 90368);   // 128*64
  unsigned short* yW2t = (unsigned short*)(w + 106752);  // 128*128

  PrepArgs pp;
  pp.zW1 = (const float*)d_in[zc[0]]; pp.zW2 = (const float*)d_in[zc[2]];
  pp.zW3 = (const float*)d_in[zc[4]];
  pp.yW1 = (const float*)d_in[yc[0]]; pp.yW2 = (const float*)d_in[yc[2]];
  pp.zW1t = zW1t; pp.zW2t = zW2t; pp.zW3t = zW3t; pp.yW1t = yW1t; pp.yW2t = yW2t;
  pp.bar = bar;

  KernelArgs mp;
  mp.x = (const float*)d_in[ix];
  mp.bm = (const float*)d_in[ibm];
  mp.zb3  = (const float*)d_in[zc[5]];
  mp.zg1  = (const float*)d_in[zc[6]]; mp.zbe1 = (const float*)d_in[zc[7]];
  mp.zg2  = (const float*)d_in[zc[8]]; mp.zbe2 = (const float*)d_in[zc[9]];
  mp.yW3v = (const float*)d_in[yc[4]]; mp.yb3 = (const float*)d_in[yc[5]];
  mp.yg1  = (const float*)d_in[yc[6]]; mp.ybe1 = (const float*)d_in[yc[7]];
  mp.yg2  = (const float*)d_in[yc[8]]; mp.ybe2 = (const float*)d_in[yc[9]];
  mp.zW1t = zW1t; mp.zW2t = zW2t; mp.zW3t = zW3t; mp.yW1t = yW1t; mp.yW2t = yW2t;
  mp.wsBN = wsBN; mp.wsLoss = wsLoss; mp.bar = bar;
  mp.out = (float*)d_out;

  bsde_prep<<<240, BLOCK, 0, stream>>>(pp);
  bsde_main<<<NUM_WG, BLOCK, LDS_BYTES, stream>>>(mp);
}
